// SimpleMambaLayer_39900246180547
// MI455X (gfx1250) — matrix operand order repaired and checked by reading, not compiled, not run on hardware
//
#include <hip/hip_runtime.h>

typedef __attribute__((ext_vector_type(16))) __bf16 v16bf;
typedef __attribute__((ext_vector_type(8)))  float  v8f;
typedef __attribute__((ext_vector_type(8)))  unsigned short us8;

__device__ __forceinline__ unsigned short f32_to_bf16(float f) {
  unsigned int u = __float_as_uint(f);
  u += 0x7FFFu + ((u >> 16) & 1u);   // round-to-nearest-even
  return (unsigned short)(u >> 16);
}

// ---------------- conversion / packing ----------------
__global__ void f2bf_kernel(const float* __restrict__ src,
                            unsigned short* __restrict__ dst, int n) {
  for (int i = blockIdx.x * blockDim.x + threadIdx.x; i < n;
       i += gridDim.x * blockDim.x)
    dst[i] = f32_to_bf16(src[i]);
}

__global__ void pack_wbc_kernel(const float* __restrict__ WB, const float* __restrict__ WC,
                                const float* __restrict__ bB, const float* __restrict__ bC,
                                unsigned short* __restrict__ WBCh, float* __restrict__ bBC) {
  int i = blockIdx.x * blockDim.x + threadIdx.x;
  if (i < 1024 * 32) {
    int e = i >> 5, c = i & 31;
    float v = (c < 16) ? WB[e * 16 + c] : WC[e * 16 + (c - 16)];
    WBCh[i] = f32_to_bf16(v);
  }
  if (i < 32) bBC[i] = (i < 16) ? bB[i] : bC[i - 16];
}

// ---------------- bf16 WMMA GEMM: C[MxN] = act(A[MxK] * B[KxN] + bias) ----------------
// BM=64, BK=32, block=128 threads (4 waves). Wave w computes rows [w*16, w*16+16)
// across NT = BN/16 column tiles. Double-buffered LDS; A tile staged with
// CDNA5 async global->LDS DMA (ASYNCcnt), B tile transposed through VGPRs.
template <int BN, int ACT>
__global__ __launch_bounds__(128) void gemm_bf16_kernel(
    const unsigned short* __restrict__ A, const unsigned short* __restrict__ Bm,
    const float* __restrict__ bias, float* __restrict__ C, int M, int N, int K) {
  (void)M;
  constexpr int NT  = BN / 16;
  constexpr int LDA = 40;   // 32 + pad (80B rows -> conflict-free, 16B aligned)
  constexpr int LDB = 40;
  constexpr int CPT = BN / 4;      // B cols per thread
  constexpr int NB8 = CPT / 8;     // us8 chunks per thread
  __shared__ unsigned short sA[2][64 * LDA];   // A tile, row-major [64][32]
  __shared__ unsigned short sB[2][BN * LDB];   // B tile, TRANSPOSED [n][k]

  const int tid  = threadIdx.x;
  const int w    = tid >> 5;
  const int lane = tid & 31;
  const int m    = lane & 15;
  const int kh   = lane >> 4;
  const int rowBase = blockIdx.x * 64;
  const int colBase = blockIdx.y * BN;

  // staging index precompute
  const int rA = tid >> 1, hfA = tid & 1;            // A: row 0..63, 16-half
  const int kkB = tid >> 2, c0B = (tid & 3) * CPT;   // B: k row, col group

  v8f acc[NT];
#pragma unroll
  for (int t = 0; t < NT; ++t) acc[t] = (v8f){0, 0, 0, 0, 0, 0, 0, 0};

  union ABu { us8 h[2]; v16bf v; };
  us8 breg[NB8];

  // async copy of this thread's 32B slice of the A tile into LDS buffer `buf`
  auto issueA = [&](int k0, int buf) {
    unsigned ldsoff = (unsigned)(size_t)&sA[buf][rA * LDA + hfA * 16];
    unsigned voff   = (unsigned)(((rowBase + rA) * K + k0 + hfA * 16) * 2);
    asm volatile("global_load_async_to_lds_b128 %0, %1, %2"
                 :: "v"(ldsoff), "v"(voff), "s"(A) : "memory");
    asm volatile("global_load_async_to_lds_b128 %0, %1, %2 offset:16"
                 :: "v"(ldsoff), "v"(voff), "s"(A) : "memory");
  };
  auto loadB = [&](int k0) {
#pragma unroll
    for (int i = 0; i < NB8; ++i)
      breg[i] = *(const us8*)(Bm + (size_t)(k0 + kkB) * N + colBase + c0B + i * 8);
  };
  auto scatterB = [&](int buf) {
#pragma unroll
    for (int i = 0; i < NB8; ++i)
#pragma unroll
      for (int j = 0; j < 8; ++j)
        sB[buf][(c0B + i * 8 + j) * LDB + kkB] = breg[i][j];
  };

  // prologue: stage tile 0
  issueA(0, 0);
  loadB(0);
  scatterB(0);
  asm volatile("s_wait_asynccnt 0x0" ::: "memory");
  __syncthreads();

  int buf = 0;
  for (int k0 = 0; k0 < K; k0 += 32) {
    const int nxt = k0 + 32;
    const bool has = nxt < K;
    if (has) {            // overlap next tile's global traffic with compute
      issueA(nxt, buf ^ 1);
      loadB(nxt);
    }

    // A fragment: lane(m,kh) holds k = {kh*8..kh*8+7} and {16+kh*8..+7}
    ABu a;
    a.h[0] = *(const us8*)&sA[buf][(w * 16 + m) * LDA + kh * 8];
    a.h[1] = *(const us8*)&sA[buf][(w * 16 + m) * LDA + 16 + kh * 8];

    // preload all B fragments, then back-to-back WMMA chain
    ABu bfr[NT];
#pragma unroll
    for (int nt = 0; nt < NT; ++nt) {
      int n = nt * 16 + m;
      bfr[nt].h[0] = *(const us8*)&sB[buf][n * LDB + kh * 8];
      bfr[nt].h[1] = *(const us8*)&sB[buf][n * LDB + 16 + kh * 8];
    }
#pragma unroll
    for (int nt = 0; nt < NT; ++nt)
      acc[nt] = __builtin_amdgcn_wmma_f32_16x16x32_bf16(
          false, a.v, false, bfr[nt].v, (short)0, acc[nt], false, false);

    if (has) {
      scatterB(buf ^ 1);
      asm volatile("s_wait_asynccnt 0x0" ::: "memory");
    }
    __syncthreads();
    buf ^= 1;
  }

  // epilogue: D layout -> row = j + kh*8, col = lane&15 (+tile offsets)
#pragma unroll
  for (int nt = 0; nt < NT; ++nt) {
    int col = colBase + nt * 16 + m;
    float bv = bias ? bias[col] : 0.0f;
#pragma unroll
    for (int j = 0; j < 8; ++j) {
      int row = rowBase + w * 16 + kh * 8 + j;
      float v = acc[nt][j] + bv;
      if (ACT == 1)  // branchless stable softplus
        v = fmaxf(v, 0.0f) + log1pf(__expf(-fabsf(v)));
      C[(size_t)row * N + col] = v;
    }
  }
}

// ---------------- selective-scan: thread per (b,e,n), shuffle-reduce over n ----------------
__global__ __launch_bounds__(256) void scan_kernel(
    const float* __restrict__ x, const float* __restrict__ dt,
    const float* __restrict__ BC, const float* __restrict__ A_log,
    const float* __restrict__ Dvec, unsigned short* __restrict__ yh) {
  int tid = blockIdx.x * 256 + threadIdx.x;   // 131072 total
  int n = tid & 15;
  int e = (tid >> 4) & 1023;
  int b = tid >> 14;

  float Ae = -__expf(A_log[e * 16 + n]);
  float Dv = Dvec[e];
  const size_t baseBE = ((size_t)b * 2048) * 1024 + e;
  const float* xp  = x  + baseBE;
  const float* dtp = dt + baseBE;
  const float* bcp = BC + (size_t)b * 2048 * 32;
  unsigned short* yp = yh + baseBE;

  float h = 0.0f;
  for (int t = 0; t < 2048; ++t) {
    float dtv = dtp[(size_t)t * 1024];        // broadcast within 16-lane group
    float xv  = xp[(size_t)t * 1024];
    float bn  = bcp[t * 32 + n];
    float cn  = bcp[t * 32 + 16 + n];
    h = __expf(Ae * dtv) * h + dtv * xv * bn;
    float part = h * cn;
    part += __shfl_xor(part, 1, 32);
    part += __shfl_xor(part, 2, 32);
    part += __shfl_xor(part, 4, 32);
    part += __shfl_xor(part, 8, 32);          // sum over n within half-wave
    if (n == 0) yp[(size_t)t * 1024] = f32_to_bf16(part + xv * Dv);
  }
}

// ---------------- LayerNorm over last dim (1024) ----------------
__global__ __launch_bounds__(256) void ln_kernel(
    const float* __restrict__ z, const float* __restrict__ gamma,
    const float* __restrict__ beta, float* __restrict__ out) {
  int row = blockIdx.x, tid = threadIdx.x;
  const float4 v = ((const float4*)(z + (size_t)row * 1024))[tid];
  float s  = v.x + v.y + v.z + v.w;
  float s2 = v.x * v.x + v.y * v.y + v.z * v.z + v.w * v.w;
#pragma unroll
  for (int off = 16; off >= 1; off >>= 1) {
    s  += __shfl_xor(s,  off, 32);
    s2 += __shfl_xor(s2, off, 32);
  }
  __shared__ float ls[8], ls2[8];
  int w = tid >> 5, lane = tid & 31;
  if (lane == 0) { ls[w] = s; ls2[w] = s2; }
  __syncthreads();
  s = 0.0f; s2 = 0.0f;
#pragma unroll
  for (int i = 0; i < 8; ++i) { s += ls[i]; s2 += ls2[i]; }
  float mu   = s * (1.0f / 1024.0f);
  float var  = s2 * (1.0f / 1024.0f) - mu * mu;
  float rstd = rsqrtf(var + 1e-5f);
  int c = tid * 4;
  float4 g  = ((const float4*)gamma)[tid];
  float4 be = ((const float4*)beta)[tid];
  float* o = out + (size_t)row * 1024 + c;
  o[0] = (v.x - mu) * rstd * g.x + be.x;
  o[1] = (v.y - mu) * rstd * g.y + be.y;
  o[2] = (v.z - mu) * rstd * g.z + be.z;
  o[3] = (v.w - mu) * rstd * g.w + be.w;
}

// ---------------- host orchestration ----------------
extern "C" void kernel_launch(void* const* d_in, const int* in_sizes, int n_in,
                              void* d_out, int out_size, void* d_ws, size_t ws_size,
                              hipStream_t stream) {
  (void)in_sizes; (void)n_in; (void)out_size; (void)ws_size;
  const float* x     = (const float*)d_in[0];
  const float* Wdt   = (const float*)d_in[1];
  const float* bdt   = (const float*)d_in[2];
  const float* WB    = (const float*)d_in[3];
  const float* bB    = (const float*)d_in[4];
  const float* WC    = (const float*)d_in[5];
  const float* bC    = (const float*)d_in[6];
  const float* Dv    = (const float*)d_in[7];
  const float* A_log = (const float*)d_in[8];
  const float* Wout  = (const float*)d_in[9];
  const float* gamma = (const float*)d_in[10];
  const float* beta  = (const float*)d_in[11];

  const size_t MT = 16384;  // B*T
  const size_t E  = 1024;
  auto al = [](size_t v) { return (v + 255) & ~(size_t)255; };
  size_t off = 0;
  size_t oXH   = off; off = al(off + MT * E * 2);   // x in bf16
  size_t oWDT  = off; off = al(off + E * E * 2);    // Wdt bf16
  size_t oWBC  = off; off = al(off + E * 32 * 2);   // [WB|WC] bf16
  size_t oWOUT = off; off = al(off + E * E * 2);    // Wout bf16
  size_t oBBC  = off; off = al(off + 32 * 4);       // [bB|bC]
  size_t oDT   = off; off = al(off + MT * E * 4);   // dt f32
  size_t oBCB  = off; off = al(off + MT * 32 * 4);  // [Bin|Cout] f32
  size_t oYH   = off; off = al(off + MT * E * 2);   // y bf16
  size_t oZB   = off; off = al(off + MT * E * 4);   // z f32

  char* ws = (char*)d_ws;
  unsigned short* xh    = (unsigned short*)(ws + oXH);
  unsigned short* wdth  = (unsigned short*)(ws + oWDT);
  unsigned short* wbch  = (unsigned short*)(ws + oWBC);
  unsigned short* wouth = (unsigned short*)(ws + oWOUT);
  float*          bbc   = (float*)(ws + oBBC);
  float*          dtb   = (float*)(ws + oDT);
  float*          bcb   = (float*)(ws + oBCB);
  unsigned short* yh    = (unsigned short*)(ws + oYH);
  float*          zb    = (float*)(ws + oZB);

  f2bf_kernel<<<4096, 256, 0, stream>>>(x, xh, (int)(MT * E));
  f2bf_kernel<<<1024, 256, 0, stream>>>(Wdt, wdth, (int)(E * E));
  f2bf_kernel<<<1024, 256, 0, stream>>>(Wout, wouth, (int)(E * E));
  pack_wbc_kernel<<<128, 256, 0, stream>>>(WB, WC, bB, bC, wbch, bbc);

  // dt = softplus(x @ Wdt + bdt)
  gemm_bf16_kernel<64, 1><<<dim3(256, 16), 128, 0, stream>>>(
      xh, wdth, bdt, dtb, 16384, 1024, 1024);
  // [Bin|Cout] = x @ [WB|WC] + [bB|bC]
  gemm_bf16_kernel<32, 0><<<dim3(256, 1), 128, 0, stream>>>(
      xh, wbch, bbc, bcb, 16384, 32, 1024);
  // sequential selective scan -> y (bf16)
  scan_kernel<<<512, 256, 0, stream>>>(x, dtb, bcb, A_log, Dv, yh);
  // z = y @ Wout
  gemm_bf16_kernel<64, 0><<<dim3(256, 16), 128, 0, stream>>>(
      yh, wouth, nullptr, zb, 16384, 1024, 1024);
  // out = LayerNorm(z) * gamma + beta
  ln_kernel<<<16384, 256, 0, stream>>>(zb, gamma, beta, (float*)d_out);
}
